// SRATBlock_41369124995839
// MI455X (gfx1250) — matrix operand order repaired and checked by reading, not compiled, not run on hardware
//
#include <hip/hip_runtime.h>
#include <hip/hip_bf16.h>

typedef __attribute__((ext_vector_type(16))) _Float16 v16h;
typedef __attribute__((ext_vector_type(8)))  float    v8f;

#define WMMA_F16(a, b, c) \
  __builtin_amdgcn_wmma_f32_16x16x32_f16(false, (a), false, (b), (short)0, (c), false, false)

// ---- problem constants -----------------------------------------------------
// B=8 H=W=128 C=192 NH=6 HD=32 WS=8 N=64 NW=256  (B*NW = 2048 windows)
#define SCALE_Q 0.17677669529663687f   // 32^-0.5
#define QKVW_OFF   0        // [576][192] f16
#define PROJW_OFF  110592   // [192][192] f16
#define FC1W_OFF   147456   // [768][192] f16
#define FC2W_OFF   294912   // [192][768] f16

// ---- WMMA tile loaders (contiguous k order k = 16*(lane/16) + elem, same in A and B) ----------
// A tile 16x32 f16, row-major source with leading dim lda.
// lane<16: M=lane,   elems 0-15 -> K=k0+0..15
// lane>=16: M=lane-16, elems 0-15 -> K=k0+16..31
__device__ __forceinline__ v16h load_A16(const _Float16* base, int lda,
                                         int m0, int k0, int lane) {
  const _Float16* p = base + (m0 + (lane & 15)) * lda + k0 + ((lane & 16) ? 16 : 0);
  v16h a;
#pragma unroll
  for (int e = 0; e < 8; ++e) a[e] = p[e];
#pragma unroll
  for (int e = 0; e < 8; ++e) a[8 + e] = p[8 + e];
  return a;
}

// B tile 32x16 where B[K][N] = W[n][k], W row-major [Nout][K] (i.e. out = A @ W^T).
// lane<16 covers K=k0+0..15, lane>=16 covers K=k0+16..31; 16 contiguous halves.
__device__ __forceinline__ v16h load_BT16(const _Float16* W, int ldw,
                                          int n0, int k0, int lane) {
  const _Float16* p = W + (n0 + (lane & 15)) * ldw + k0 + ((lane & 16) ? 16 : 0);
  v16h b;
#pragma unroll
  for (int e = 0; e < 16; ++e) b[e] = p[e];
  return b;
}

// B tile 32x16 from a K-major (row = K) source: B[K][N] = M[k][n], ldm = row stride.
__device__ __forceinline__ v16h load_B16(const _Float16* M, int ldm,
                                         int k0, int n0, int lane) {
  int n = n0 + (lane & 15);
  int kb = k0 + ((lane & 16) ? 16 : 0);
  v16h b;
#pragma unroll
  for (int e = 0; e < 16; ++e) b[e] = M[(kb + e) * ldm + n];
  return b;
}

__device__ __forceinline__ int region3(int p) {  // Swin shift-mask region label
  return (p < 120) ? 0 : ((p < 124) ? 1 : 2);
}

// ---- K0: convert fp32 weights -> f16 workspace -----------------------------
__global__ void wconv_kernel(const float* __restrict__ qkvw,
                             const float* __restrict__ projw,
                             const float* __restrict__ fc1w,
                             const float* __restrict__ fc2w,
                             _Float16* __restrict__ wbuf) {
  int i = blockIdx.x * 256 + threadIdx.x;
  if (i < 110592) wbuf[QKVW_OFF + i] = (_Float16)qkvw[i];
  if (i < 36864)  wbuf[PROJW_OFF + i] = (_Float16)projw[i];
  if (i < 147456) {
    wbuf[FC1W_OFF + i] = (_Float16)fc1w[i];
    wbuf[FC2W_OFF + i] = (_Float16)fc2w[i];
  }
}

// ---- K1: fused LN1 + shift/partition + QKV + attention + proj + residual ---
// One block per window (2048 blocks, 8 waves). Dynamic LDS:
//   Ash/Osh : 64*192 f16 (24 KB, reused)   Qsh : 64*576 f16 (72 KB)
//   Psh     : 6*64*64 f16 (48 KB)          total 147456 B (CDNA5 320KB WGP LDS)
__global__ void __launch_bounds__(256) swin_attn_kernel(
    const float* __restrict__ x,
    const float* __restrict__ n1g, const float* __restrict__ n1b,
    const float* __restrict__ qkv_b, const float* __restrict__ proj_b,
    const float* __restrict__ rpb,
    const _Float16* __restrict__ wbuf,
    float* __restrict__ out) {
  extern __shared__ _Float16 smem[];
  _Float16* Ash = smem;                 // 64 x 192 (LN'd window tokens)
  _Float16* Qsh = smem + 64 * 192;      // 64 x 576 (q|k|v, q pre-scaled)
  _Float16* Psh = Qsh + 64 * 576;       // 6 x 64 x 64 (softmax probs)
  _Float16* Osh = Ash;                  // reuse: 64 x 192 attention output

  const int blk = blockIdx.x;
  const int b   = blk >> 8;
  const int ww  = blk & 255;
  const int wi  = ww >> 4, wj = ww & 15;
  const int tid = threadIdx.x, wave = tid >> 5, lane = tid & 31;

  // ---- phase 1: gather (cyclic shift) + LayerNorm1 -> Ash (f16) ----
  const float* xb = x + (size_t)b * 16384 * 192;
#pragma unroll 1
  for (int t = wave * 8; t < wave * 8 + 8; ++t) {
    int r = t >> 3, c = t & 7;
    int h0 = (wi * 8 + r + 4) & 127;  // roll(-SHIFT): shifted[p] = orig[p+4]
    int w0 = (wj * 8 + c + 4) & 127;
    const float* row = xb + (size_t)(h0 * 128 + w0) * 192;
    float v[6], s = 0.f, s2 = 0.f;
#pragma unroll
    for (int j = 0; j < 6; ++j) {
      v[j] = row[lane + 32 * j];
      s += v[j]; s2 += v[j] * v[j];
    }
#pragma unroll
    for (int m = 1; m < 32; m <<= 1) {
      s  += __shfl_xor(s,  m, 32);
      s2 += __shfl_xor(s2, m, 32);
    }
    float mu = s * (1.f / 192.f);
    float rs = rsqrtf(s2 * (1.f / 192.f) - mu * mu + 1e-5f);
#pragma unroll
    for (int j = 0; j < 6; ++j) {
      int cc = lane + 32 * j;
      Ash[t * 192 + cc] = (_Float16)(((v[j] - mu) * rs) * n1g[cc] + n1b[cc]);
    }
  }
  __syncthreads();

  // ---- phase 2: QKV GEMM (64x192 @ 192x576^T) -> Qsh ----
  const _Float16* qkvW = wbuf + QKVW_OFF;  // [576][192]
#pragma unroll 1
  for (int it = 0; it < 18; ++it) {
    int tile = wave * 18 + it;          // 4 M-tiles x 36 N-tiles
    int m0 = (tile / 36) * 16, n0 = (tile % 36) * 16;
    v8f acc = {};
#pragma unroll
    for (int kt = 0; kt < 6; ++kt) {
      v16h a  = load_A16(Ash, 192, m0, kt * 32, lane);
      v16h bm = load_BT16(qkvW, 192, n0, kt * 32, lane);
      acc = WMMA_F16(a, bm, acc);
    }
    int n  = n0 + (lane & 15);
    int mb = m0 + ((lane & 16) ? 8 : 0);
    float bias = qkv_b[n];
    float scl  = (n < 192) ? SCALE_Q : 1.0f;   // q * HD^-0.5
#pragma unroll
    for (int rr = 0; rr < 8; ++rr)
      Qsh[(mb + rr) * 576 + n] = (_Float16)((acc[rr] + bias) * scl);
  }
  __syncthreads();

  // ---- phase 3: per-head attention (wave h handles head h) ----
  if (wave < 6) {
    const int h = wave;
    const _Float16* qh = Qsh + h * 32;
    const _Float16* kh = Qsh + 192 + h * 32;
    const _Float16* vh = Qsh + 384 + h * 32;
    _Float16* Ph = Psh + h * 64 * 64;

    // S = q @ k^T : 16 tiles, K=HD=32 -> one WMMA each
    v8f acc[4][4];
#pragma unroll
    for (int mt = 0; mt < 4; ++mt) {
      v16h a = load_A16(qh, 576, mt * 16, 0, lane);
#pragma unroll
      for (int nt = 0; nt < 4; ++nt) {
        v16h bm = load_BT16(kh, 576, nt * 16, 0, lane);
        v8f z = {};
        acc[mt][nt] = WMMA_F16(a, bm, z);
      }
    }

    // rel-pos bias + shift mask + row softmax, entirely in registers.
    // C-layout: row m lives in one 16-lane half => shfl_xor(1,2,4,8) reduces it.
    const int nk_lane = lane & 15;
    const int half8   = (lane & 16) ? 8 : 0;
#pragma unroll
    for (int mt = 0; mt < 4; ++mt) {
#pragma unroll
      for (int rr = 0; rr < 8; ++rr) {
        int mq = mt * 16 + rr + half8;
        int iq = mq >> 3, jq = mq & 7;
        int lab_q = region3(wi * 8 + iq) * 3 + region3(wj * 8 + jq);
        float vals[4];
#pragma unroll
        for (int nt = 0; nt < 4; ++nt) {
          int nk = nt * 16 + nk_lane;
          int ik = nk >> 3, jk = nk & 7;
          int idx = (iq - ik + 7) * 15 + (jq - jk + 7);
          int lab_k = region3(wi * 8 + ik) * 3 + region3(wj * 8 + jk);
          float msk = (lab_q != lab_k) ? -100.f : 0.f;
          vals[nt] = acc[mt][nt][rr] + rpb[idx * 6 + h] + msk;
        }
        float mx = fmaxf(fmaxf(vals[0], vals[1]), fmaxf(vals[2], vals[3]));
#pragma unroll
        for (int m = 1; m < 16; m <<= 1) mx = fmaxf(mx, __shfl_xor(mx, m, 32));
        float sm = 0.f;
#pragma unroll
        for (int nt = 0; nt < 4; ++nt) { vals[nt] = __expf(vals[nt] - mx); sm += vals[nt]; }
#pragma unroll
        for (int m = 1; m < 16; m <<= 1) sm += __shfl_xor(sm, m, 32);
        float inv = 1.f / sm;
#pragma unroll
        for (int nt = 0; nt < 4; ++nt)
          Ph[mq * 64 + nt * 16 + nk_lane] = (_Float16)(vals[nt] * inv);
      }
    }

    // O = P @ V : 64x64 @ 64x32, K=64 -> 2 WMMA per 16x16 tile
#pragma unroll
    for (int mt = 0; mt < 4; ++mt) {
#pragma unroll
      for (int dn = 0; dn < 2; ++dn) {
        v8f o = {};
#pragma unroll
        for (int kt = 0; kt < 2; ++kt) {
          v16h a  = load_A16(Ph, 64, mt * 16, kt * 32, lane);
          v16h bm = load_B16(vh, 576, kt * 32, dn * 16, lane);
          o = WMMA_F16(a, bm, o);
        }
        int n  = dn * 16 + (lane & 15);
        int mb = mt * 16 + half8;
#pragma unroll
        for (int rr = 0; rr < 8; ++rr)
          Osh[(mb + rr) * 192 + h * 32 + n] = (_Float16)o[rr];
      }
    }
  }
  __syncthreads();

  // ---- phase 4: proj GEMM + window-reverse/unshift + residual -> out ----
  const _Float16* projW = wbuf + PROJW_OFF;  // [192][192]
#pragma unroll 1
  for (int it = 0; it < 6; ++it) {
    int tile = wave * 6 + it;          // 4 x 12 tiles
    int m0 = (tile / 12) * 16, n0 = (tile % 12) * 16;
    v8f acc = {};
#pragma unroll
    for (int kt = 0; kt < 6; ++kt) {
      v16h a  = load_A16(Osh, 192, m0, kt * 32, lane);
      v16h bm = load_BT16(projW, 192, n0, kt * 32, lane);
      acc = WMMA_F16(a, bm, acc);
    }
    int n  = n0 + (lane & 15);
    int mb = m0 + ((lane & 16) ? 8 : 0);
    float pb = proj_b[n];
#pragma unroll
    for (int rr = 0; rr < 8; ++rr) {
      int t = mb + rr;
      int hh = (wi * 8 + (t >> 3) + 4) & 127;  // roll(+SHIFT) == same src index
      int wwp = (wj * 8 + (t & 7) + 4) & 127;
      size_t rowg = (size_t)b * 16384 + hh * 128 + wwp;
      out[rowg * 192 + n] = x[rowg * 192 + n] + acc[rr] + pb;  // x1 = x + attn
    }
  }
}

// ---- K2: fused LN2 + FC1 + leaky-relu + FC2 + residual ---------------------
// One block per 64 rows (2048 blocks). LDS: Ash 24 KB + Hsh 96 KB = 122880 B.
__global__ void __launch_bounds__(256) swin_mlp_kernel(
    const float* __restrict__ n2g, const float* __restrict__ n2b,
    const float* __restrict__ fc1_b, const float* __restrict__ fc2_b,
    const _Float16* __restrict__ wbuf,
    float* __restrict__ out) {    // holds x1 on entry, final result on exit
  extern __shared__ _Float16 smem[];
  _Float16* Ash = smem;             // 64 x 192 (LN2'd rows)
  _Float16* Hsh = smem + 64 * 192;  // 64 x 768 (hidden, f16)

  const size_t row0 = (size_t)blockIdx.x * 64;
  const int tid = threadIdx.x, wave = tid >> 5, lane = tid & 31;

  // LN2 -> Ash
#pragma unroll 1
  for (int t = wave * 8; t < wave * 8 + 8; ++t) {
    const float* row = out + (row0 + t) * 192;
    float v[6], s = 0.f, s2 = 0.f;
#pragma unroll
    for (int j = 0; j < 6; ++j) {
      v[j] = row[lane + 32 * j];
      s += v[j]; s2 += v[j] * v[j];
    }
#pragma unroll
    for (int m = 1; m < 32; m <<= 1) {
      s  += __shfl_xor(s,  m, 32);
      s2 += __shfl_xor(s2, m, 32);
    }
    float mu = s * (1.f / 192.f);
    float rs = rsqrtf(s2 * (1.f / 192.f) - mu * mu + 1e-5f);
#pragma unroll
    for (int j = 0; j < 6; ++j) {
      int cc = lane + 32 * j;
      Ash[t * 192 + cc] = (_Float16)(((v[j] - mu) * rs) * n2g[cc] + n2b[cc]);
    }
  }
  __syncthreads();

  // FC1 (64x192 @ 192x768^T) + leaky relu -> Hsh
  const _Float16* fc1W = wbuf + FC1W_OFF;  // [768][192]
#pragma unroll 1
  for (int it = 0; it < 24; ++it) {
    int tile = wave * 24 + it;         // 4 x 48 tiles
    int m0 = (tile / 48) * 16, n0 = (tile % 48) * 16;
    v8f acc = {};
#pragma unroll
    for (int kt = 0; kt < 6; ++kt) {
      v16h a  = load_A16(Ash, 192, m0, kt * 32, lane);
      v16h bm = load_BT16(fc1W, 192, n0, kt * 32, lane);
      acc = WMMA_F16(a, bm, acc);
    }
    int n  = n0 + (lane & 15);
    int mb = m0 + ((lane & 16) ? 8 : 0);
    float bias = fc1_b[n];
#pragma unroll
    for (int rr = 0; rr < 8; ++rr) {
      float vv = acc[rr] + bias;
      vv = (vv > 0.f) ? vv : 0.2f * vv;        // leaky_relu 0.2
      Hsh[(mb + rr) * 768 + n] = (_Float16)vv;
    }
  }
  __syncthreads();

  // FC2 (64x768 @ 768x192^T) + residual -> out
  const _Float16* fc2W = wbuf + FC2W_OFF;  // [192][768]
#pragma unroll 1
  for (int it = 0; it < 6; ++it) {
    int tile = wave * 6 + it;          // 4 x 12 tiles
    int m0 = (tile / 12) * 16, n0 = (tile % 12) * 16;
    v8f acc = {};
#pragma unroll
    for (int kt = 0; kt < 24; ++kt) {
      v16h a  = load_A16(Hsh, 768, m0, kt * 32, lane);
      v16h bm = load_BT16(fc2W, 768, n0, kt * 32, lane);
      acc = WMMA_F16(a, bm, acc);
    }
    int n  = n0 + (lane & 15);
    int mb = m0 + ((lane & 16) ? 8 : 0);
    float bias = fc2_b[n];
#pragma unroll
    for (int rr = 0; rr < 8; ++rr) {
      size_t r = row0 + mb + rr;
      out[r * 192 + n] += acc[rr] + bias;      // x1 + mlp
    }
  }
}

extern "C" void kernel_launch(void* const* d_in, const int* in_sizes, int n_in,
                              void* d_out, int out_size, void* d_ws, size_t ws_size,
                              hipStream_t stream) {
  const float* x      = (const float*)d_in[0];
  const float* n1g    = (const float*)d_in[1];
  const float* n1b    = (const float*)d_in[2];
  const float* qkv_w  = (const float*)d_in[3];
  const float* qkv_b  = (const float*)d_in[4];
  const float* proj_w = (const float*)d_in[5];
  const float* proj_b = (const float*)d_in[6];
  const float* rpb    = (const float*)d_in[7];
  const float* n2g    = (const float*)d_in[8];
  const float* n2b    = (const float*)d_in[9];
  const float* fc1_w  = (const float*)d_in[10];
  const float* fc1_b  = (const float*)d_in[11];
  const float* fc2_w  = (const float*)d_in[12];
  const float* fc2_b  = (const float*)d_in[13];
  float* out = (float*)d_out;
  _Float16* wbuf = (_Float16*)d_ws;   // 442368 f16 = 884736 B of weights

  wconv_kernel<<<576, 256, 0, stream>>>(qkv_w, proj_w, fc1_w, fc2_w, wbuf);

  size_t lds_attn = (size_t)(64 * 192 + 64 * 576 + 6 * 64 * 64) * sizeof(_Float16); // 147456
  swin_attn_kernel<<<2048, 256, lds_attn, stream>>>(
      x, n1g, n1b, qkv_b, proj_b, rpb, wbuf, out);

  size_t lds_mlp = (size_t)(64 * 192 + 64 * 768) * sizeof(_Float16);                // 122880
  swin_mlp_kernel<<<2048, 256, lds_mlp, stream>>>(
      n2g, n2b, fc1_b, fc2_b, wbuf, out);
}
